// TFBlock_48163763257807
// MI455X (gfx1250) — compile-verified, matrix operand order checked by reading, not run on hardware
//
#include <hip/hip_runtime.h>
#include <hip/hip_bf16.h>

typedef __attribute__((ext_vector_type(16))) _Float16 v16h;
typedef __attribute__((ext_vector_type(8)))  _Float16 v8h;
typedef __attribute__((ext_vector_type(8)))  float    v8f;
typedef int v4i __attribute__((__vector_size__(4 * sizeof(int))));

#define F16 _Float16

// ---------------------------------------------------------------- constants
static constexpr int Bsz = 2, Sn = 1024, Dm = 768, Hn = 12, DH = 64, Mf = 64;
static constexpr int DFF = 4 * Dm;          // 3072
static constexpr int ROWS = Bsz * Sn;       // 2048
static constexpr int D3 = 3 * Dm;           // 2304

// ---------------------------------------------------------------- async copy
#if defined(__has_builtin)
#if __has_builtin(__builtin_amdgcn_global_load_async_to_lds_b128)
#define HAVE_ASYNC_BUILTIN 1
#endif
#if __has_builtin(__builtin_amdgcn_s_wait_asynccnt)
#define HAVE_WAIT_BUILTIN 1
#endif
#endif

typedef __attribute__((address_space(1))) v4i* gptr_v4i;
typedef __attribute__((address_space(3))) v4i* lptr_v4i;

__device__ __forceinline__ void async_cp16(const F16* g, F16* l) {
#if defined(HAVE_ASYNC_BUILTIN)
    __builtin_amdgcn_global_load_async_to_lds_b128(
        (gptr_v4i)g, (lptr_v4i)l, 0, 0);
#else
    unsigned lofs = (unsigned)(unsigned long long)
        (__attribute__((address_space(3))) F16*)l;
    asm volatile("global_load_async_to_lds_b128 %0, %1, off"
                 :: "v"(lofs), "v"(g) : "memory");
#endif
}

__device__ __forceinline__ void wait_async0() {
#if defined(HAVE_WAIT_BUILTIN)
    __builtin_amdgcn_s_wait_asynccnt(0);
#else
    asm volatile("s_wait_asynccnt 0" ::: "memory");
#endif
}

// ---------------------------------------------------------------- helpers
__device__ __forceinline__ v8f wmma16(v16h a, v16h b, v8f c) {
    return __builtin_amdgcn_wmma_f32_16x16x32_f16(false, a, false, b,
                                                  (short)0, c, false, false);
}

// Load a 16x32 f16 fragment (A-layout; B-layout identical when matrix is
// stored [n][k]) from LDS, row-major, stride in halfs.
__device__ __forceinline__ v16h ld_frag(const F16* base, int row0, int stride,
                                        int k0, int lane) {
    int r  = lane & 15;
    int ks = (lane & 16) ? 8 : 0;
    const F16* p = base + (row0 + r) * stride + k0 + ks;
    v8h lo = *(const v8h*)(p);
    v8h hi = *(const v8h*)(p + 16);
    v16h o;
#pragma unroll
    for (int i = 0; i < 8; ++i) { o[i] = lo[i]; o[i + 8] = hi[i]; }
    return o;
}

// ---------------------------------------------------------------- transpose+cast
// w: [K][N] f32 row-major  ->  wt: [N][K] f16 row-major
__global__ __launch_bounds__(256) void transpose_cast(const float* __restrict__ w,
                                                      F16* __restrict__ wt,
                                                      int K, int N) {
    __shared__ float tile[32][33];
    int n0 = blockIdx.x * 32, k0 = blockIdx.y * 32;
    int tx = threadIdx.x, ty = threadIdx.y;
    for (int r = ty; r < 32; r += 8)
        tile[r][tx] = w[(size_t)(k0 + r) * N + n0 + tx];
    __syncthreads();
    for (int r = ty; r < 32; r += 8)
        wt[(size_t)(n0 + r) * K + k0 + tx] = (F16)tile[tx][r];
}

// ---------------------------------------------------------------- layernorm -> f16
__global__ __launch_bounds__(256) void ln_kernel(const float* __restrict__ in,
                                                 const float* __restrict__ g,
                                                 const float* __restrict__ b,
                                                 F16* __restrict__ out) {
    __shared__ float s1[256], s2[256];
    int row = blockIdx.x, t = threadIdx.x;
    const float* xr = in + (size_t)row * Dm;
    float v0 = xr[t], v1 = xr[t + 256], v2 = xr[t + 512];
    s1[t] = v0 + v1 + v2;
    s2[t] = v0 * v0 + v1 * v1 + v2 * v2;
    __syncthreads();
    for (int off = 128; off > 0; off >>= 1) {
        if (t < off) { s1[t] += s1[t + off]; s2[t] += s2[t + off]; }
        __syncthreads();
    }
    float mu = s1[0] * (1.0f / Dm);
    float var = s2[0] * (1.0f / Dm) - mu * mu;
    float rs = rsqrtf(var + 1e-5f);
    F16* orow = out + (size_t)row * Dm;
    orow[t]       = (F16)((v0 - mu) * rs * g[t]       + b[t]);
    orow[t + 256] = (F16)((v1 - mu) * rs * g[t + 256] + b[t + 256]);
    orow[t + 512] = (F16)((v2 - mu) * rs * g[t + 512] + b[t + 512]);
}

// ---------------------------------------------------------------- WMMA GEMM
// C[M][N] = A[M][K](f16) * BT[N][K](f16)^T + bias
// Block tile 128x256, 8 waves (2x4), 64x64 per wave, BK=32, double-buffered
// LDS fed by async global->LDS copies.
// EPI 0: f32 store, EPI 1: f32 store + residual, EPI 2: gelu(tanh) -> f16
template <int EPI>
__global__ __launch_bounds__(256) void wmma_gemm(const F16* __restrict__ A,
                                                 const F16* __restrict__ BT,
                                                 const float* __restrict__ bias,
                                                 const float* __restrict__ res,
                                                 void* __restrict__ outp,
                                                 int N, int K) {
    constexpr int LS = 40;                       // padded LDS stride (halfs)
    __shared__ __align__(16) F16 As[2][128 * LS];
    __shared__ __align__(16) F16 Bs[2][256 * LS];

    const int tid = threadIdx.x, lane = tid & 31, wid = tid >> 5;
    const int wm = wid >> 2, wn = wid & 3;       // 2 x 4 wave grid
    const int bm = blockIdx.y, bn = blockIdx.x;

    const int ar = tid >> 2;                     // 0..63
    const int ao = (tid & 3) * 8;                // 0,8,16,24 (halfs)

    const F16* Ag = A  + (size_t)(bm * 128) * K;
    const F16* Bg = BT + (size_t)(bn * 256) * K;

    v8f acc[4][4];
#pragma unroll
    for (int i = 0; i < 4; ++i)
#pragma unroll
        for (int j = 0; j < 4; ++j)
#pragma unroll
            for (int e = 0; e < 8; ++e) acc[i][j][e] = 0.0f;

    const int KT = K >> 5;

    // ---- stage tile `ko` into buffer `buf` (each lane: one b128 per row-group)
    auto stage = [&](int ko, int buf) {
#pragma unroll
        for (int j = 0; j < 2; ++j)
            async_cp16(Ag + (size_t)(ar + 64 * j) * K + ko + ao,
                       &As[buf][(ar + 64 * j) * LS + ao]);
#pragma unroll
        for (int j = 0; j < 4; ++j)
            async_cp16(Bg + (size_t)(ar + 64 * j) * K + ko + ao,
                       &Bs[buf][(ar + 64 * j) * LS + ao]);
    };

    stage(0, 0);
    for (int kt = 0; kt < KT; ++kt) {
        wait_async0();                 // our async writes into buf done
        __syncthreads();               // everyone's writes visible
        const int buf = kt & 1;
        if (kt + 1 < KT) stage((kt + 1) * 32, buf ^ 1);

        v16h af[4], bf[4];
#pragma unroll
        for (int ms = 0; ms < 4; ++ms)
            af[ms] = ld_frag(&As[buf][0], wm * 64 + ms * 16, LS, 0, lane);
#pragma unroll
        for (int ns = 0; ns < 4; ++ns)
            bf[ns] = ld_frag(&Bs[buf][0], wn * 64 + ns * 16, LS, 0, lane);
#pragma unroll
        for (int ms = 0; ms < 4; ++ms)
#pragma unroll
            for (int ns = 0; ns < 4; ++ns)
                acc[ms][ns] = wmma16(af[ms], bf[ns], acc[ms][ns]);
    }

    const int hi8 = (lane & 16) ? 8 : 0;
    const int nl = lane & 15;
    const int rowb = bm * 128 + wm * 64;
    const int colb = bn * 256 + wn * 64;
#pragma unroll
    for (int ms = 0; ms < 4; ++ms)
#pragma unroll
        for (int ns = 0; ns < 4; ++ns)
#pragma unroll
            for (int i = 0; i < 8; ++i) {
                int row = rowb + ms * 16 + i + hi8;
                int col = colb + ns * 16 + nl;
                float v = acc[ms][ns][i] + bias[col];
                size_t idx = (size_t)row * N + col;
                if (EPI == 0) {
                    ((float*)outp)[idx] = v;
                } else if (EPI == 1) {
                    ((float*)outp)[idx] = v + res[idx];
                } else {
                    float c = v + 0.044715f * v * v * v;
                    float gl = 0.5f * v * (1.0f + tanhf(0.7978845608028654f * c));
                    ((F16*)outp)[idx] = (F16)gl;
                }
            }
}

// ---------------------------------------------------------------- FAVOR+ features
// qkv f32 [B][S][3D] -> phi_q/phi_k f16 [B*H][S][M], v f16 [B*H][S][DH]
__global__ __launch_bounds__(128) void phi_kernel(const float* __restrict__ qkv,
                                                  const float* __restrict__ wfeat,
                                                  F16* __restrict__ phiq,
                                                  F16* __restrict__ phik,
                                                  F16* __restrict__ vf) {
    __shared__ float sq_[64], sk_[64];
    const int s = blockIdx.x, bh = blockIdx.y;
    const int b = bh / Hn, h = bh % Hn;
    const int t = threadIdx.x;
    const float scale = 0.35355339059327373f;   // 64^-0.25
    const float* row = qkv + (size_t)(b * Sn + s) * D3 + h * DH;
    if (t < 64) {
        sq_[t] = row[t] * scale;
        vf[((size_t)bh * Sn + s) * DH + t] = (F16)row[2 * Dm + t];
    } else {
        sk_[t - 64] = row[Dm + (t - 64)] * scale;
    }
    __syncthreads();
    const int m = t & 63;
    const float* wr = wfeat + m * DH;
    const float* xv = (t < 64) ? sq_ : sk_;
    float proj = 0.0f, ss = 0.0f;
#pragma unroll 8
    for (int d = 0; d < DH; ++d) { proj += xv[d] * wr[d]; ss += xv[d] * xv[d]; }
    float ph = expf(proj - 0.5f * ss) * 0.125f + 1e-6f;
    F16* dst = (t < 64) ? phiq : phik;
    dst[((size_t)bh * Sn + s) * Mf + m] = (F16)ph;
}

// ---------------------------------------------------------------- chunked causal linear attention
// One workgroup (4 waves) per (b,h). Chunks of 64 rows. State Z' is
// [M=64][DH+16=80] f32 in wave accumulators; column 64 of V' is all-ones so
// the denominator cumsum rides along as column 64 of Z'/num.
__global__ __launch_bounds__(128) void performer_attn(const F16* __restrict__ phiq,
                                                      const F16* __restrict__ phik,
                                                      const F16* __restrict__ vf,
                                                      F16* __restrict__ attn) {
    constexpr int LS = 72;                       // padded LDS stride (halfs)
    __shared__ __align__(16) F16 sQ[64 * LS];    // [s][m]
    __shared__ __align__(16) F16 sK[64 * LS];    // [s][m]   (B for Phiq*Phik^T)
    __shared__ __align__(16) F16 sKT[64 * LS];   // [m][s]   (A for Z update)
    __shared__ __align__(16) F16 sVT[80 * LS];   // [d'][s]  (B for *V')
    __shared__ __align__(16) F16 sZT[80 * LS];   // [d'][m]  (B for Phiq*Z')
    __shared__ __align__(16) F16 sAm[64 * LS];   // masked scores [i][j]
    __shared__ float sDen[64];

    const int bh = blockIdx.x;
    const int b = bh / Hn, h = bh % Hn;
    const int tid = threadIdx.x, lane = tid & 31, w = tid >> 5;
    const int hi8 = (lane & 16) ? 8 : 0;
    const int nl = lane & 15;

    v8f zacc[5];
#pragma unroll
    for (int nt = 0; nt < 5; ++nt)
#pragma unroll
        for (int e = 0; e < 8; ++e) zacc[nt][e] = 0.0f;

    for (int i = tid; i < 80 * LS; i += 128) sZT[i] = (F16)0.0f;

    const F16* pq = phiq + (size_t)bh * Sn * Mf;
    const F16* pk = phik + (size_t)bh * Sn * Mf;
    const F16* pv = vf + (size_t)bh * Sn * DH;

    for (int c = 0; c < 16; ++c) {
        const int s0 = c * 64;
        // ---- stage chunk tiles
#pragma unroll
        for (int q = 0; q < 4; ++q) {
            int cc = tid + 128 * q;
            int row = cc >> 3, off = (cc & 7) * 8;
            v8h vq = *(const v8h*)(pq + ((size_t)(s0 + row)) * Mf + off);
            *(v8h*)&sQ[row * LS + off] = vq;
            v8h vk = *(const v8h*)(pk + ((size_t)(s0 + row)) * Mf + off);
            *(v8h*)&sK[row * LS + off] = vk;
#pragma unroll
            for (int e = 0; e < 8; ++e) sKT[(off + e) * LS + row] = vk[e];
            v8h vv = *(const v8h*)(pv + ((size_t)(s0 + row)) * DH + off);
#pragma unroll
            for (int e = 0; e < 8; ++e) sVT[(off + e) * LS + row] = vv[e];
        }
        for (int i = tid; i < 16 * 64; i += 128) {
            int r = 64 + (i >> 6), col = i & 63;
            sVT[r * LS + col] = (r == 64) ? (F16)1.0f : (F16)0.0f;
        }
        __syncthreads();                         // b1

        v16h aq0 = ld_frag(sQ, w * 16, LS, 0, lane);
        v16h aq1 = ld_frag(sQ, w * 16, LS, 32, lane);

        // num' = Phiq @ Z'   (previous chunk's state)
        v8f nacc[5];
#pragma unroll
        for (int nt = 0; nt < 5; ++nt) {
            v8f z;
#pragma unroll
            for (int e = 0; e < 8; ++e) z[e] = 0.0f;
            v16h bz0 = ld_frag(sZT, nt * 16, LS, 0, lane);
            v16h bz1 = ld_frag(sZT, nt * 16, LS, 32, lane);
            z = wmma16(aq0, bz0, z);
            nacc[nt] = wmma16(aq1, bz1, z);
        }
        // scores = Phiq @ Phik^T, causal mask, spill to LDS as f16
#pragma unroll
        for (int nt = 0; nt < 4; ++nt) {
            v8f sc;
#pragma unroll
            for (int e = 0; e < 8; ++e) sc[e] = 0.0f;
            v16h bk0 = ld_frag(sK, nt * 16, LS, 0, lane);
            v16h bk1 = ld_frag(sK, nt * 16, LS, 32, lane);
            sc = wmma16(aq0, bk0, sc);
            sc = wmma16(aq1, bk1, sc);
#pragma unroll
            for (int i = 0; i < 8; ++i) {
                int ig = w * 16 + i + hi8;
                int j = nt * 16 + nl;
                sAm[ig * LS + j] = (F16)((j <= ig) ? sc[i] : 0.0f);
            }
        }
        __syncthreads();                         // b2

        // num' += mask(scores) @ V'   and   Z' += Phik^T @ V'
        v16h am0 = ld_frag(sAm, w * 16, LS, 0, lane);
        v16h am1 = ld_frag(sAm, w * 16, LS, 32, lane);
        v16h ak0 = ld_frag(sKT, w * 16, LS, 0, lane);
        v16h ak1 = ld_frag(sKT, w * 16, LS, 32, lane);
#pragma unroll
        for (int nt = 0; nt < 5; ++nt) {
            v16h bv0 = ld_frag(sVT, nt * 16, LS, 0, lane);
            v16h bv1 = ld_frag(sVT, nt * 16, LS, 32, lane);
            nacc[nt] = wmma16(am0, bv0, nacc[nt]);
            nacc[nt] = wmma16(am1, bv1, nacc[nt]);
            zacc[nt] = wmma16(ak0, bv0, zacc[nt]);
            zacc[nt] = wmma16(ak1, bv1, zacc[nt]);
        }
        // publish new state (f16, transposed) + denominator column
#pragma unroll
        for (int nt = 0; nt < 5; ++nt)
#pragma unroll
            for (int i = 0; i < 8; ++i) {
                int m = w * 16 + i + hi8;
                int d = nt * 16 + nl;
                sZT[d * LS + m] = (F16)zacc[nt][i];
            }
        if (nl == 0) {
#pragma unroll
            for (int i = 0; i < 8; ++i) sDen[w * 16 + i + hi8] = nacc[4][i];
        }
        __syncthreads();                         // b3

        // out = num / den, merge heads: attn[b][s][h*64+d]
#pragma unroll
        for (int nt = 0; nt < 4; ++nt)
#pragma unroll
            for (int i = 0; i < 8; ++i) {
                int sl = w * 16 + i + hi8;
                int d = nt * 16 + nl;
                float o = nacc[nt][i] / sDen[sl];
                attn[((size_t)(b * Sn + s0 + sl)) * Dm + h * DH + d] = (F16)o;
            }
    }
}

// ---------------------------------------------------------------- launch
extern "C" void kernel_launch(void* const* d_in, const int* in_sizes, int n_in,
                              void* d_out, int out_size, void* d_ws, size_t ws_size,
                              hipStream_t stream) {
    const float* x      = (const float*)d_in[0];
    const float* ln1_g  = (const float*)d_in[1];
    const float* ln1_b  = (const float*)d_in[2];
    const float* w_attn = (const float*)d_in[3];
    const float* b_attn = (const float*)d_in[4];
    const float* w_feat = (const float*)d_in[5];
    const float* w_proj = (const float*)d_in[6];
    const float* b_proj = (const float*)d_in[7];
    const float* ln2_g  = (const float*)d_in[8];
    const float* ln2_b  = (const float*)d_in[9];
    const float* w_fc   = (const float*)d_in[10];
    const float* b_fc   = (const float*)d_in[11];
    const float* w_out  = (const float*)d_in[12];
    const float* b_out  = (const float*)d_in[13];

    char* ws = (char*)d_ws;
    size_t off = 0;
    F16*  wattnT = (F16*)(ws + off); off += (size_t)D3 * Dm * 2;
    F16*  wprojT = (F16*)(ws + off); off += (size_t)Dm * Dm * 2;
    F16*  wfcT   = (F16*)(ws + off); off += (size_t)DFF * Dm * 2;
    F16*  woutT  = (F16*)(ws + off); off += (size_t)Dm * DFF * 2;
    F16*  a16    = (F16*)(ws + off); off += (size_t)ROWS * Dm * 2;     // ln1/ln2 out
    float* qkv   = (float*)(ws + off); off += (size_t)ROWS * D3 * 4;   // qkv / hidden
    F16*  phiq   = (F16*)(ws + off); off += (size_t)Bsz * Hn * Sn * Mf * 2;
    F16*  phik   = (F16*)(ws + off); off += (size_t)Bsz * Hn * Sn * Mf * 2;
    F16*  v16    = (F16*)(ws + off); off += (size_t)Bsz * Hn * Sn * DH * 2;
    F16*  attn16 = (F16*)(ws + off); off += (size_t)ROWS * Dm * 2;
    float* x1    = (float*)(ws + off); off += (size_t)ROWS * Dm * 4;
    F16*  m16 = a16;                 // alias: a16 dead after qkv GEMM
    F16*  h16 = (F16*)qkv;           // alias: qkv dead after phi kernel

    dim3 tb(32, 8);
    transpose_cast<<<dim3(D3 / 32, Dm / 32), tb, 0, stream>>>(w_attn, wattnT, Dm, D3);
    transpose_cast<<<dim3(Dm / 32, Dm / 32), tb, 0, stream>>>(w_proj, wprojT, Dm, Dm);
    transpose_cast<<<dim3(DFF / 32, Dm / 32), tb, 0, stream>>>(w_fc, wfcT, Dm, DFF);
    transpose_cast<<<dim3(Dm / 32, DFF / 32), tb, 0, stream>>>(w_out, woutT, DFF, Dm);

    ln_kernel<<<ROWS, 256, 0, stream>>>(x, ln1_g, ln1_b, a16);

    wmma_gemm<0><<<dim3(D3 / 256, ROWS / 128), 256, 0, stream>>>(
        a16, wattnT, b_attn, nullptr, qkv, D3, Dm);

    phi_kernel<<<dim3(Sn, Bsz * Hn), 128, 0, stream>>>(qkv, w_feat, phiq, phik, v16);

    performer_attn<<<Bsz * Hn, 128, 0, stream>>>(phiq, phik, v16, attn16);

    wmma_gemm<1><<<dim3(Dm / 256, ROWS / 128), 256, 0, stream>>>(
        attn16, wprojT, b_proj, x, x1, Dm, Dm);

    ln_kernel<<<ROWS, 256, 0, stream>>>(x1, ln2_g, ln2_b, m16);

    wmma_gemm<2><<<dim3(DFF / 256, ROWS / 128), 256, 0, stream>>>(
        m16, wfcT, b_fc, nullptr, h16, DFF, Dm);

    wmma_gemm<1><<<dim3(Dm / 256, ROWS / 128), 256, 0, stream>>>(
        h16, woutT, b_out, x1, d_out, Dm, DFF);
}
